// LSTMModel_19636590477938
// MI455X (gfx1250) — matrix operand order repaired and checked by reading, not compiled, not run on hardware
//
#include <hip/hip_runtime.h>

// ---------------------------------------------------------------------------
// Masked autoregressive LSTMCell for MI455X (gfx1250, wave32, WMMA).
// B=1024, F=64, S=512, H=128.  64 blocks x 256 threads (8 waves).
// Each block owns 16 batch rows for the whole sequence; weights (f16),
// h/c state, gates and feedback buffers are LDS-resident (~283 KB of the
// 320 KB WGP LDS). Matrix work uses v_wmma_f32_16x16x32_f16.
// ---------------------------------------------------------------------------

typedef __attribute__((ext_vector_type(16))) _Float16 v16h;
typedef __attribute__((ext_vector_type(8)))  _Float16 v8h;
typedef __attribute__((ext_vector_type(8)))  float    v8f;

#define B_TOT    1024
#define F_DIM    64
#define S_DIM    512
#define H_DIM    128
#define G_DIM    512        // 4*H
#define MT       16         // batch rows per block (one WMMA M-tile)
#define NTHREADS 256        // 8 wave32

// ---- LDS byte offsets (all 16B aligned) -----------------------------------
#define OFF_WIH    0                       // 512*64*2   = 65536  (f16 W_ih)
#define OFF_WHH    (OFF_WIH  + 65536)      // 512*128*2  = 131072 (f16 W_hh)
#define OFF_WDEN   (OFF_WHH  + 131072)     // 64*128*2   = 16384  (f16 W_den)
#define OFF_BIASG  (OFF_WDEN + 16384)      // 512*4      = 2048   (b_ih+b_hh)
#define OFF_BIASD  (OFF_BIASG+ 2048)       // 64*4       = 256
#define OFF_AIN    (OFF_BIASD+ 256)        // 16*64*2    = 2048   (f16 input tile)
#define OFF_AH     (OFF_AIN  + 2048)       // 16*128*2   = 4096   (f16 h tile)
#define OFF_GATES  (OFF_AH   + 4096)       // 16*512*4   = 32768  (f32 gates)
#define OFF_PRED   (OFF_GATES+ 32768)      // 16*64*4    = 4096
#define OFF_INP    (OFF_PRED + 4096)       // 16*64*4    = 4096   (f32 inp_t)
#define OFF_OPREV  (OFF_INP  + 4096)       // 16*64*4    = 4096   (prev out_t)
#define OFF_CST    (OFF_OPREV+ 4096)       // 16*128*4   = 8192   (c state)
#define OFF_HST    (OFF_CST  + 8192)       // 16*128*4   = 8192   (h state)
#define OFF_LENS   (OFF_HST  + 8192)       // 2*16*4     = 128
#define LDS_BYTES  (OFF_LENS + 128)        // = 283008 bytes

__device__ __forceinline__ float sigmoidf_fast(float x) {
  return 1.0f / (1.0f + __expf(-x));
}
__device__ __forceinline__ float tanhf_fast(float x) {
  float e = __expf(-2.0f * x);
  return (1.0f - e) / (1.0f + e);
}

// A-matrix fragment (16x32 f16, row-major in LDS, leading dim `ld` halves):
// lane l (l<16): row M=l, K in {kt+0..15}
// lane l (>=16): row M=l-16, K in {kt+16..31}
__device__ __forceinline__ v16h ld_fragA(const _Float16* base, int ld, int kt) {
  const int lane = threadIdx.x & 31;
  const int row  = lane & 15;
  const int k0   = kt + ((lane >> 4) << 4);
  const _Float16* p = base + row * ld + k0;
  v8h lo = *(const v8h*)(p);
  v8h hi = *(const v8h*)(p + 8);
  v16h r;
#pragma unroll
  for (int i = 0; i < 8; ++i) { r[i] = lo[i]; r[8 + i] = hi[i]; }
  return r;
}

// B-matrix fragment (32x16 f16). B[k][n] = W[n][k] for C = A * W^T, W
// row-major in LDS with leading dim `ld`.  Same K order as ld_fragA: lanes 0-15 hold column
// N=lane, K=kt+0..15 (contiguous); lanes 16-31 hold N=lane-16, K=kt+16..31.
__device__ __forceinline__ v16h ld_fragB(const _Float16* w, int n_base, int ld, int kt) {
  const int lane = threadIdx.x & 31;
  const int n    = n_base + (lane & 15);
  const int k0   = kt + ((lane >> 4) << 4);
  return *(const v16h*)(w + n * ld + k0);   // 32B contiguous -> 2x ds_load_b128
}

__global__ __launch_bounds__(NTHREADS)
void lstm_ar_wmma_kernel(const float* __restrict__ x,
                         const int*   __restrict__ alen_g,
                         const int*   __restrict__ plen_g,
                         const float* __restrict__ Wih,
                         const float* __restrict__ Whh,
                         const float* __restrict__ bih,
                         const float* __restrict__ bhh,
                         const float* __restrict__ Wden,
                         const float* __restrict__ bden,
                         const float* __restrict__ initch,
                         float*       __restrict__ out) {
  extern __shared__ char smem[];
  _Float16* sWih  = (_Float16*)(smem + OFF_WIH);
  _Float16* sWhh  = (_Float16*)(smem + OFF_WHH);
  _Float16* sWden = (_Float16*)(smem + OFF_WDEN);
  float*    sBiasG= (float*)(smem + OFF_BIASG);
  float*    sBiasD= (float*)(smem + OFF_BIASD);
  _Float16* sAin  = (_Float16*)(smem + OFF_AIN);
  _Float16* sAh   = (_Float16*)(smem + OFF_AH);
  float*    sGates= (float*)(smem + OFF_GATES);
  float*    sPred = (float*)(smem + OFF_PRED);
  float*    sInp  = (float*)(smem + OFF_INP);
  float*    sOPrev= (float*)(smem + OFF_OPREV);
  float*    sC    = (float*)(smem + OFF_CST);
  float*    sH    = (float*)(smem + OFF_HST);
  int*      sPlen = (int*)(smem + OFF_LENS);
  int*      sAlen = (int*)(smem + OFF_LENS + 64);

  const int tid  = threadIdx.x;
  const int wave = tid >> 5;
  const int lane = tid & 31;
  const int lcol = lane & 15;      // N index within a tile
  const int grp  = lane >> 4;      // C-layout row group (M = grp*8 + j)
  const int m0   = blockIdx.x * MT;

  // ------------------ one-time setup: weights -> f16 LDS -------------------
  for (int i = tid; i < G_DIM * F_DIM; i += NTHREADS) sWih[i]  = (_Float16)Wih[i];
  for (int i = tid; i < G_DIM * H_DIM; i += NTHREADS) sWhh[i]  = (_Float16)Whh[i];
  for (int i = tid; i < F_DIM * H_DIM; i += NTHREADS) sWden[i] = (_Float16)Wden[i];
  for (int i = tid; i < G_DIM; i += NTHREADS) sBiasG[i] = bih[i] + bhh[i];
  if (tid < F_DIM) sBiasD[tid] = bden[tid];
  if (tid < MT) { sPlen[tid] = plen_g[m0 + tid]; sAlen[tid] = alen_g[m0 + tid]; }
  for (int i = tid; i < MT * H_DIM; i += NTHREADS) {
    int hc = i & (H_DIM - 1);
    float c0 = initch[hc];
    float h0 = initch[H_DIM + hc];
    sC[i] = c0; sH[i] = h0; sAh[i] = (_Float16)h0;
  }
  for (int i = tid; i < MT * F_DIM; i += NTHREADS) sOPrev[i] = 0.0f;
  __syncthreads();

  // --------------------------- recurrence over t ---------------------------
  for (int t = 0; t < S_DIM; ++t) {
    // Phase I: build inp_t (autoregressive selection), f32 + f16 copies.
    {
      const int base = tid * 4;            // 16*64 / 256 = 4 elems/thread
      const int r    = base >> 6;
      const int f    = base & 63;
      const bool use_pred = (t >= sAlen[r]) && (t < sPlen[r]);
      float v0, v1, v2, v3;
      if (use_pred) {
        const float4 p = *(const float4*)(sOPrev + r * F_DIM + f);
        v0 = p.x; v1 = p.y; v2 = p.z; v3 = p.w;
      } else {
        const float* xp = x + ((size_t)(m0 + r) * F_DIM + f) * S_DIM + t;
        v0 = xp[0]; v1 = xp[S_DIM]; v2 = xp[2 * S_DIM]; v3 = xp[3 * S_DIM];
        if (t + 16 < S_DIM) __builtin_prefetch(xp + 16, 0, 0);  // global_prefetch_b8
      }
      *(float4*)(sInp + r * F_DIM + f) = make_float4(v0, v1, v2, v3);
      _Float16* ai = sAin + r * F_DIM + f;
      ai[0] = (_Float16)v0; ai[1] = (_Float16)v1;
      ai[2] = (_Float16)v2; ai[3] = (_Float16)v3;
    }
    __syncthreads();

    // Phase G: gates = inp @ W_ih^T + h @ W_hh^T   (16 x 512, K=64 + K=128).
    // 32 N-tiles across 8 waves -> 4 tiles * 6 WMMAs per wave.
#pragma unroll
    for (int nt = 0; nt < 4; ++nt) {
      const int n_base = (wave * 4 + nt) * 16;
      v8f acc = {};
#pragma unroll
      for (int kt = 0; kt < F_DIM; kt += 32) {
        v16h a = ld_fragA(sAin, F_DIM, kt);
        v16h b = ld_fragB(sWih, n_base, F_DIM, kt);
        acc = __builtin_amdgcn_wmma_f32_16x16x32_f16(false, a, false, b,
                                                     (short)0, acc, false, false);
      }
#pragma unroll
      for (int kt = 0; kt < H_DIM; kt += 32) {
        v16h a = ld_fragA(sAh, H_DIM, kt);
        v16h b = ld_fragB(sWhh, n_base, H_DIM, kt);
        acc = __builtin_amdgcn_wmma_f32_16x16x32_f16(false, a, false, b,
                                                     (short)0, acc, false, false);
      }
#pragma unroll
      for (int j = 0; j < 8; ++j)
        sGates[(grp * 8 + j) * G_DIM + n_base + lcol] = acc[j];
    }
    __syncthreads();

    // Phase E: LSTM pointwise with masked state update (2048 elems / 256 thr).
#pragma unroll
    for (int e = 0; e < 8; ++e) {
      const int idx = tid * 8 + e;
      const int r   = idx >> 7;
      const int hc  = idx & (H_DIM - 1);
      const bool sel = t < sPlen[r];
      const float gi = sGates[r * G_DIM + hc]             + sBiasG[hc];
      const float gf = sGates[r * G_DIM + H_DIM + hc]     + sBiasG[H_DIM + hc];
      const float gg = sGates[r * G_DIM + 2 * H_DIM + hc] + sBiasG[2 * H_DIM + hc];
      const float go = sGates[r * G_DIM + 3 * H_DIM + hc] + sBiasG[3 * H_DIM + hc];
      const float c_old = sC[r * H_DIM + hc];
      const float h_old = sH[r * H_DIM + hc];
      const float cn = sigmoidf_fast(gf) * c_old + sigmoidf_fast(gi) * tanhf_fast(gg);
      const float hn = sigmoidf_fast(go) * tanhf_fast(cn);
      const float c_out = sel ? cn : c_old;
      const float h_out = sel ? hn : h_old;
      sC[r * H_DIM + hc]  = c_out;
      sH[r * H_DIM + hc]  = h_out;
      sAh[r * H_DIM + hc] = (_Float16)h_out;   // feeds next-step GEMM + head
    }
    __syncthreads();

    // Phase P: dense head pred = h_new @ W_den^T + b_den (16x64, K=128).
    // Waves 0-3 take one 16-column tile each (wave-uniform branch, EXEC full).
    if (wave < 4) {
      const int n_base = wave * 16;
      v8f acc = {};
#pragma unroll
      for (int kt = 0; kt < H_DIM; kt += 32) {
        v16h a = ld_fragA(sAh, H_DIM, kt);
        v16h b = ld_fragB(sWden, n_base, H_DIM, kt);
        acc = __builtin_amdgcn_wmma_f32_16x16x32_f16(false, a, false, b,
                                                     (short)0, acc, false, false);
      }
      const float bd = sBiasD[n_base + lcol];
#pragma unroll
      for (int j = 0; j < 8; ++j)
        sPred[(grp * 8 + j) * F_DIM + n_base + lcol] = acc[j] + bd;
    }
    __syncthreads();

    // Phase O: out_t = sel ? pred : inp_t ; store to ys[t] and keep as prev.
    {
      const int base = tid * 4;
      const int r    = base >> 6;
      const int f    = base & 63;
      const bool sel = t < sPlen[r];
      const float4 pv = *(const float4*)(sPred + r * F_DIM + f);
      const float4 iv = *(const float4*)(sInp  + r * F_DIM + f);
      const float4 ov = sel ? pv : iv;
      *(float4*)(sOPrev + r * F_DIM + f) = ov;
      *(float4*)(out + (size_t)t * (B_TOT * F_DIM) + (size_t)(m0 + r) * F_DIM + f) = ov;
    }
    __syncthreads();
  }
}

extern "C" void kernel_launch(void* const* d_in, const int* in_sizes, int n_in,
                              void* d_out, int out_size, void* d_ws, size_t ws_size,
                              hipStream_t stream) {
  (void)in_sizes; (void)n_in; (void)d_ws; (void)ws_size; (void)out_size;
  const float* x      = (const float*)d_in[0];
  const int*   alen   = (const int*)  d_in[1];
  const int*   plen   = (const int*)  d_in[2];
  const float* Wih    = (const float*)d_in[3];
  const float* Whh    = (const float*)d_in[4];
  const float* bih    = (const float*)d_in[5];
  const float* bhh    = (const float*)d_in[6];
  const float* Wden   = (const float*)d_in[7];
  const float* bden   = (const float*)d_in[8];
  const float* initch = (const float*)d_in[9];
  float*       out    = (float*)d_out;

  dim3 grid(B_TOT / MT);   // 64 blocks, one per 16-row batch tile
  dim3 block(NTHREADS);    // 8 wave32
  hipLaunchKernelGGL(lstm_ar_wmma_kernel, grid, block, LDS_BYTES, stream,
                     x, alen, plen, Wih, Whh, bih, bhh, Wden, bden, initch, out);
}
